// TTT_23656679866733
// MI455X (gfx1250) — matrix operand order repaired and checked by reading, not compiled, not run on hardware
//
#include <hip/hip_runtime.h>
#include <hip/hip_bf16.h>
#include <math.h>

// MI455X / gfx1250 implementation.
// Workspace requirement: 16*2304*2496*4 + 16*2304*832*4 ~= 491 MB in d_ws.
//
// All WMMA operands are staged in LDS in *fragment-ready* layout (the exact
// per-lane register image of v_wmma_f32_16x16x32_bf16 operands): fragment
// loads are one 32B LDS read (2x ds_load_b128); staging is global_load_b128
// + v_cvt_pk_bf16_f32 + ds_store_b128/b32.

typedef __attribute__((ext_vector_type(16))) __bf16   v16bf;
typedef __attribute__((ext_vector_type(2)))  __bf16   v2bf;
typedef __attribute__((ext_vector_type(8)))  float    v8f;
typedef __attribute__((ext_vector_type(8)))  unsigned v8u;

#define QKVW  2496
#define CDIM  768
#define DDIM  64
#define NSEQ  2304
#define BSZ   16
#define NHEAD 12
#define XCATW 832
#define HWDIM 48
#define SCALE 0.3333333333333333f   // 9^-0.5

// Packs to one v_cvt_pk_bf16_f32: lo -> [15:0], hi -> [31:16].
__device__ __forceinline__ unsigned pk_bf16(float lo, float hi) {
  v2bf t;
  t[0] = (__bf16)lo;
  t[1] = (__bf16)hi;
  return __builtin_bit_cast(unsigned, t);
}

// One fragment = 32 lanes x 8 dwords (32B per lane), lane-major.
__device__ __forceinline__ v16bf load_frag(const v8u* frags, int f) {
  return __builtin_bit_cast(v16bf, frags[f * 32 + (threadIdx.x & 31)]);
}

__device__ __forceinline__ v8f wmma_bf16(v16bf a, v16bf b, v8f c) {
  return __builtin_amdgcn_wmma_f32_16x16x32_bf16(false, a, false, b, (short)0, c,
                                                 false, false);
}

// Fragment-slot addressing (dword index):
//   A frag (16x32, MxK): lane = (m&15) + 16*hi; k<16: hi=k>>3, p=(k&7)>>1
//                                        k>=16: hi=(k-16)>>3, p=4+((k-16)&7)>>1
//   B frag (32x16, KxN): lane = 16*hi + n, same hi/p as A; kA/w1f/w2f use hi=(k&31)>>4, p=(k&15)>>1

// ---------------------------------------------------------------------------
// C[M,Nc] = A[M,K] @ Wt[Nc,K]^T + bias.  Block tile 128x64, K step 32.
// 8 waves; each wave owns a 32x32 output (4 accumulators, 4 WMMA / K-step).
// ---------------------------------------------------------------------------
__global__ __launch_bounds__(256)
void gemm_bias_kernel(const float* __restrict__ A, const float* __restrict__ Wt,
                      const float* __restrict__ bias, float* __restrict__ C,
                      int M, int Nc, int K) {
  __shared__ v8u Af[8 * 32];   // 8 A fragments (m-groups), 8KB
  __shared__ v8u Bf[4 * 32];   // 4 B fragments (n-groups), 4KB
  const int tid = threadIdx.x, wid = tid >> 5, lane = tid & 31;
  const int m0 = blockIdx.y * 128, n0 = blockIdx.x * 64;
  const int fm = (wid >> 1) * 2;   // wave's A-fragment pair
  const int fn = (wid & 1) * 2;    // wave's B-fragment pair

  unsigned* Au = (unsigned*)Af;
  unsigned* Bu = (unsigned*)Bf;
  // A staging: two tasks/thread over (row 0..127, k-seg of 8)
  const int ar  = tid >> 2;            // rows tid>>2 and +64
  const int seg = (tid & 3) * 8;
  unsigned* adst0 = Au + (((ar >> 4) * 32 + ((seg >> 3) & 1) * 16 + (ar & 15)) * 8 +
                          ((seg & 16) ? 4 : 0));
  unsigned* adst1 = adst0 + 4 * 32 * 8;  // row + 64 => frag + 4
  // B staging: one task/thread over (col 0..63, k-seg of 8)
  unsigned* bdst = Bu + (((ar >> 4) * 32 + ((seg >> 3) & 1) * 16 + (ar & 15)) * 8 +
                         ((seg & 16) ? 4 : 0));

  v8f c00 = {}, c01 = {}, c10 = {}, c11 = {};
  for (int k0 = 0; k0 < K; k0 += 32) {
    const float* ap0 = A  + (size_t)(m0 + ar) * K + k0 + seg;
    const float* ap1 = ap0 + (size_t)64 * K;
    const float* bp  = Wt + (size_t)(n0 + ar) * K + k0 + seg;
    const float4 a0 = *(const float4*)ap0;
    const float4 a1 = *(const float4*)(ap0 + 4);
    const float4 a2 = *(const float4*)ap1;
    const float4 a3 = *(const float4*)(ap1 + 4);
    const float4 w0 = *(const float4*)bp;
    const float4 w1v = *(const float4*)(bp + 4);
    uint4 p0, p1, pb;
    p0.x = pk_bf16(a0.x, a0.y);  p0.y = pk_bf16(a0.z, a0.w);
    p0.z = pk_bf16(a1.x, a1.y);  p0.w = pk_bf16(a1.z, a1.w);
    p1.x = pk_bf16(a2.x, a2.y);  p1.y = pk_bf16(a2.z, a2.w);
    p1.z = pk_bf16(a3.x, a3.y);  p1.w = pk_bf16(a3.z, a3.w);
    pb.x = pk_bf16(w0.x, w0.y);  pb.y = pk_bf16(w0.z, w0.w);
    pb.z = pk_bf16(w1v.x, w1v.y); pb.w = pk_bf16(w1v.z, w1v.w);
    *(uint4*)adst0 = p0;
    *(uint4*)adst1 = p1;
    *(uint4*)bdst  = pb;
    __syncthreads();
    if (k0 + 32 < K)
      __builtin_prefetch(ap0 + 32, 0, 0);   // global_prefetch_b8, next K panel
    v16bf am0 = load_frag(Af, fm);
    v16bf am1 = load_frag(Af, fm + 1);
    v16bf bn0 = load_frag(Bf, fn);
    v16bf bn1 = load_frag(Bf, fn + 1);
    c00 = wmma_bf16(am0, bn0, c00);
    c01 = wmma_bf16(am0, bn1, c01);
    c10 = wmma_bf16(am1, bn0, c10);
    c11 = wmma_bf16(am1, bn1, c11);
    __syncthreads();
  }
  const int nl = lane & 15, hi = lane >> 4;
  const int colA = n0 + fn * 16 + nl;
  const int colB = n0 + (fn + 1) * 16 + nl;
  const float biasA = bias[colA], biasB = bias[colB];
#pragma unroll
  for (int i = 0; i < 8; ++i) {
    const int row0 = m0 + fm * 16 + i + hi * 8;
    const int row1 = row0 + 16;
    C[(size_t)row0 * Nc + colA] = c00[i] + biasA;
    C[(size_t)row0 * Nc + colB] = c01[i] + biasB;
    C[(size_t)row1 * Nc + colA] = c10[i] + biasA;
    C[(size_t)row1 * Nc + colB] = c11[i] + biasB;
  }
}

// ---------------------------------------------------------------------------
// Fused per-(b,h) swiglu TTT update. One block per (b,h); 8 waves.
// g is accumulated transposed (gT = E^T @ k): the E tile in WMMA C-layout
// registers maps to its own lane's A-fragment slot -> one ds_store_b128.
// ---------------------------------------------------------------------------
struct SwigluOverlay {
  union {
    struct {
      v8u kA[4 * 32];   // k/q as A frags: f = (nrow>>4)*2 + (d>>5)
      v8u kB[4 * 32];   // k as B frags (K = n): f = d>>4
      v8u eA[4 * 32];   // e*a as E^T A frags: f = e>>4
      v8u gA[4 * 32];   // e*dz as E^T A frags
    } t;
    float gred[2 * 4096];  // gT spill: [2][e][d]
  };
};

__global__ __launch_bounds__(256)
void ttt_swiglu_kernel(const float* __restrict__ qkv,
                       const float* __restrict__ w1,
                       const float* __restrict__ w2,
                       float* __restrict__ xcat) {
  __shared__ v8u w1f[8 * 32];   // w1 B frags: f = (din>>5)*4 + (e>>4), 8KB
  __shared__ v8u w2f[8 * 32];
  __shared__ SwigluOverlay u;   // 32KB
  __shared__ float norms[128];

  const int tid = threadIdx.x, wid = tid >> 5, lane = tid & 31;
  const int b = blockIdx.x / NHEAD, h = blockIdx.x % NHEAD;
  const int nl = lane & 15, hi = lane >> 4;

  const int zm = wid >> 2, zn = wid & 3;          // z tiles: 2x4 of 16x16
  const int egrp = wid >> 1, dg0 = (wid & 1) * 2; // gT tiles: 4x4, 2 per wave

  const size_t rowbase = (size_t)b * NSEQ;
  const int kcol = CDIM + h * DDIM;
  const int vcol = 2 * CDIM + h * DDIM;
  const int qcol = h * DDIM;

  // ---- stage w1/w2 -> B-fragment images (one-time) ----
  {
    const float* w1p = w1 + (size_t)h * 4096;
    const float* w2p = w2 + (size_t)h * 4096;
    unsigned* w1u_ = (unsigned*)w1f;
    unsigned* w2u_ = (unsigned*)w2f;
    for (int q = tid; q < 2048; q += 256) {
      const int d0 = (q >> 6) * 2;  // even d_in
      const int e  = q & 63;
      const int dst = ((((d0 >> 5) << 2) + (e >> 4)) * 32 +
                       (((d0 & 31) >= 16) ? 16 : 0) + (e & 15)) * 8 +
                      ((d0 & 15) >> 1);
      w1u_[dst] = pk_bf16(w1p[d0 * 64 + e], w1p[(d0 + 1) * 64 + e]);
      w2u_[dst] = pk_bf16(w2p[d0 * 64 + e], w2p[(d0 + 1) * 64 + e]);
    }
  }

  v8f g1a = {}, g1b = {}, g2a = {}, g2b = {};

  // Staging geometry (A-image): thread -> (row 0..31, d-segment of 8)
  const int ar  = tid >> 3;
  const int aseg = (tid & 7) * 8;
  unsigned* kAdst = (unsigned*)u.t.kA +
      ((((ar >> 4) * 2 + (aseg >> 5)) * 32 + ((aseg & 16) ? 16 : 0) + (ar & 15)) * 8 +
       ((aseg >> 3) & 1) * 4);
  // Staging geometry (B-image): thread -> (n pair 0..15, d-segment of 4)
  const int np   = tid >> 4;
  const int dseg = (tid & 15) * 4;

  __syncthreads();

  // ================= pass 1: z1/z2, gates, gT accumulation =================
  for (int n0 = 0; n0 < NSEQ; n0 += 32) {
    {  // k tile -> A-image (for z = k @ w)
      const float* src = qkv + (rowbase + n0 + ar) * QKVW + kcol + aseg;
      const float4 x0 = *(const float4*)src;
      const float4 x1 = *(const float4*)(src + 4);
      uint4 pkd;
      pkd.x = pk_bf16(x0.x, x0.y); pkd.y = pk_bf16(x0.z, x0.w);
      pkd.z = pk_bf16(x1.x, x1.y); pkd.w = pk_bf16(x1.z, x1.w);
      *(uint4*)kAdst = pkd;
    }
    {  // k tile -> B-image (for gT = E^T @ k); pairs across consecutive n
      const float* r0 = qkv + (rowbase + n0 + 2 * np) * QKVW + kcol + dseg;
      const float4 x0 = *(const float4*)r0;
      const float4 x1 = *(const float4*)(r0 + QKVW);
      const float a0[4] = {x0.x, x0.y, x0.z, x0.w};
      const float a1[4] = {x1.x, x1.y, x1.z, x1.w};
      unsigned* kb = (unsigned*)u.t.kB;
      const int lhi = (np & 4) ? 16 : 0;
#pragma unroll
      for (int di = 0; di < 4; ++di) {
        const int d = dseg + di;
        kb[((d >> 4) * 32 + lhi + (d & 15)) * 8 + (((np >> 1) & 4) | (np & 3))] = pk_bf16(a0[di], a1[di]);
      }
    }
    __syncthreads();

    // z1/z2 16x16 tile per wave
    v8f z1 = {}, z2 = {};
#pragma unroll
    for (int kk = 0; kk < 2; ++kk) {
      v16bf a = load_frag(u.t.kA, zm * 2 + kk);
      z1 = wmma_bf16(a, load_frag(w1f, kk * 4 + zn), z1);
      z2 = wmma_bf16(a, load_frag(w2f, kk * 4 + zn), z2);
    }

    // gates -> E tiles, written directly into E^T A-fragment slots
    float ea[8], eg[8];
#pragma unroll
    for (int i = 0; i < 8; ++i) {
      const int rloc = zm * 16 + 8 * hi + i;   // n within tile
      const int cloc = zn * 16 + nl;           // e
      const float v  = qkv[(rowbase + n0 + rloc) * QKVW + vcol + cloc];
      const float ee = -v * (SCALE / (float)NSEQ);
      const float s  = 1.0f / (1.0f + __expf(-z2[i]));
      ea[i] = ee * (z2[i] * s);
      eg[i] = ee * z1[i] * (s * (1.0f + z2[i] * (1.0f - s)));
    }
    {
      uint4 ew, gw;
      ew.x = pk_bf16(ea[0], ea[1]); ew.y = pk_bf16(ea[2], ea[3]);
      ew.z = pk_bf16(ea[4], ea[5]); ew.w = pk_bf16(ea[6], ea[7]);
      gw.x = pk_bf16(eg[0], eg[1]); gw.y = pk_bf16(eg[2], eg[3]);
      gw.z = pk_bf16(eg[4], eg[5]); gw.w = pk_bf16(eg[6], eg[7]);
      const int ebase = (zn * 32 + lane) * 8 + zm * 4;  // own lane's slot
      *(uint4*)((unsigned*)u.t.eA + ebase) = ew;
      *(uint4*)((unsigned*)u.t.gA + ebase) = gw;
    }
    __syncthreads();

    // gT += E^T @ k  (A = E^T frags, B = k B-frags)
    v16bf aE  = load_frag(u.t.eA, egrp);
    v16bf aG  = load_frag(u.t.gA, egrp);
    v16bf bk0 = load_frag(u.t.kB, dg0);
    v16bf bk1 = load_frag(u.t.kB, dg0 + 1);
    g1a = wmma_bf16(aE, bk0, g1a);
    g1b = wmma_bf16(aE, bk1, g1b);
    g2a = wmma_bf16(aG, bk0, g2a);
    g2b = wmma_bf16(aG, bk1, g2b);
    __syncthreads();
  }

  // ---- spill gT [e][d] and reduce: norm over d == contiguous row sum ----
#pragma unroll
  for (int i = 0; i < 8; ++i) {
    const int e = egrp * 16 + 8 * hi + i;
    u.gred[e * 64 + dg0 * 16 + nl]              = g1a[i];
    u.gred[e * 64 + (dg0 + 1) * 16 + nl]        = g1b[i];
    u.gred[4096 + e * 64 + dg0 * 16 + nl]       = g2a[i];
    u.gred[4096 + e * 64 + (dg0 + 1) * 16 + nl] = g2b[i];
  }
  __syncthreads();

  if (tid < 128) {
    const int gi = tid >> 6, e = tid & 63;
    const float* row = u.gred + gi * 4096 + e * 64;
    float ss = 0.0f;
    for (int d = 0; d < 64; ++d) ss += row[d] * row[d];
    norms[tid] = 1.0f / (sqrtf(ss) + 1.0f);
  }
  __syncthreads();

  // ---- w{1,2}u = w - gT^T / (||.||+1), in-place on the B-frag images ----
  {
    __bf16* w1h = (__bf16*)w1f;
    __bf16* w2h = (__bf16*)w2f;
    for (int idx = tid; idx < 4096; idx += 256) {
      const int d = idx >> 6, e = idx & 63;
      const int h16 = (((((d >> 5) << 2) + (e >> 4)) * 32 +
                        (((d & 31) >= 16) ? 16 : 0) + (e & 15)) * 8 +
                       ((d & 15) >> 1)) * 2 + (d & 1);
      w1h[h16] = (__bf16)((float)w1h[h16] - u.gred[e * 64 + d] * norms[e]);
      w2h[h16] = (__bf16)((float)w2h[h16] - u.gred[4096 + e * 64 + d] * norms[64 + e]);
    }
  }
  __syncthreads();

  // ================= pass 2: x1 = (q @ w1u) * silu(q @ w2u) =================
  for (int n0 = 0; n0 < NSEQ; n0 += 32) {
    {  // q tile -> A-image
      const float* src = qkv + (rowbase + n0 + ar) * QKVW + qcol + aseg;
      const float4 x0 = *(const float4*)src;
      const float4 x1 = *(const float4*)(src + 4);
      uint4 pkd;
      pkd.x = pk_bf16(x0.x, x0.y); pkd.y = pk_bf16(x0.z, x0.w);
      pkd.z = pk_bf16(x1.x, x1.y); pkd.w = pk_bf16(x1.z, x1.w);
      *(uint4*)kAdst = pkd;
    }
    __syncthreads();
    v8f y1 = {}, y2 = {};
#pragma unroll
    for (int kk = 0; kk < 2; ++kk) {
      v16bf a = load_frag(u.t.kA, zm * 2 + kk);
      y1 = wmma_bf16(a, load_frag(w1f, kk * 4 + zn), y1);
      y2 = wmma_bf16(a, load_frag(w2f, kk * 4 + zn), y2);
    }
#pragma unroll
    for (int i = 0; i < 8; ++i) {
      const int rloc = zm * 16 + 8 * hi + i;
      const int cloc = zn * 16 + nl;
      const float s = 1.0f / (1.0f + __expf(-y2[i]));
      xcat[(rowbase + n0 + rloc) * XCATW + h * DDIM + cloc] = y1[i] * (y2[i] * s);
    }
    __syncthreads();
  }
}

// ---------------------------------------------------------------------------
// 3x3 depthwise TTT branch. One block per (b, c) channel over a 48x48 grid.
// ---------------------------------------------------------------------------
__global__ __launch_bounds__(256)
void ttt_dwconv_kernel(const float* __restrict__ qkv,
                       const float* __restrict__ w3,
                       float* __restrict__ xcat) {
  __shared__ float dots[9];
  __shared__ float w3s[9];
  const int tid = threadIdx.x;
  const int b = blockIdx.x >> 6;
  const int c = blockIdx.x & 63;
  const size_t rowbase = (size_t)b * NSEQ;
  const int q2c = 3 * CDIM + c;
  const int k2c = 3 * CDIM + DDIM + c;
  const int v2c = 3 * CDIM + 2 * DDIM + c;

  if (tid < 9) dots[tid] = 0.0f;
  __syncthreads();

  float part[9];
#pragma unroll
  for (int j = 0; j < 9; ++j) part[j] = 0.0f;

  for (int i = tid; i < NSEQ; i += 256) {
    const int y = i / HWDIM, x = i - y * HWDIM;
    const float ev = -qkv[(rowbase + i) * QKVW + v2c] * (SCALE / (float)NSEQ);
#pragma unroll
    for (int dy = -1; dy <= 1; ++dy) {
#pragma unroll
      for (int dx = -1; dx <= 1; ++dx) {
        const int yy = y + dy, xx = x + dx;
        float kv = 0.0f;
        if (yy >= 0 && yy < HWDIM && xx >= 0 && xx < HWDIM)
          kv = qkv[(rowbase + yy * HWDIM + xx) * QKVW + k2c];
        part[(dy + 1) * 3 + (dx + 1)] += kv * ev;
      }
    }
  }
#pragma unroll
  for (int j = 0; j < 9; ++j) atomicAdd(&dots[j], part[j]);
  __syncthreads();

  if (tid == 0) {
    float ss = 0.0f;
#pragma unroll
    for (int j = 0; j < 9; ++j) ss += dots[j] * dots[j];
    const float inv = 1.0f / (sqrtf(ss) + 1.0f);
#pragma unroll
    for (int j = 0; j < 9; ++j) w3s[j] = w3[c * 9 + j] - dots[j] * inv;
  }
  __syncthreads();

  for (int i = tid; i < NSEQ; i += 256) {
    const int y = i / HWDIM, x = i - y * HWDIM;
    float acc = 0.0f;
#pragma unroll
    for (int dy = -1; dy <= 1; ++dy) {
#pragma unroll
      for (int dx = -1; dx <= 1; ++dx) {
        const int yy = y + dy, xx = x + dx;
        if (yy >= 0 && yy < HWDIM && xx >= 0 && xx < HWDIM)
          acc += w3s[(dy + 1) * 3 + (dx + 1)] *
                 qkv[(rowbase + yy * HWDIM + xx) * QKVW + q2c];
      }
    }
    xcat[(rowbase + i) * XCATW + CDIM + c] = acc;
  }
}

// ---------------------------------------------------------------------------
extern "C" void kernel_launch(void* const* d_in, const int* in_sizes, int n_in,
                              void* d_out, int out_size, void* d_ws, size_t ws_size,
                              hipStream_t stream) {
  const float* x      = (const float*)d_in[0];
  const float* qkv_w  = (const float*)d_in[1];
  const float* qkv_b  = (const float*)d_in[2];
  const float* w1     = (const float*)d_in[3];
  const float* w2     = (const float*)d_in[4];
  const float* w3     = (const float*)d_in[5];
  const float* proj_w = (const float*)d_in[6];
  const float* proj_b = (const float*)d_in[7];
  float* out = (float*)d_out;

  float* qkv  = (float*)d_ws;                          // [B*N, 2496]
  float* xcat = qkv + (size_t)BSZ * NSEQ * QKVW;       // [B*N, 832]

  const int M = BSZ * NSEQ;                            // 36864
  dim3 blk(256);

  // 1) qkv = x @ qkv_w^T + qkv_b
  gemm_bias_kernel<<<dim3(QKVW / 64, M / 128), blk, 0, stream>>>(
      x, qkv_w, qkv_b, qkv, M, QKVW, CDIM);

  // 2) swiglu TTT branch -> xcat[:, :768]
  ttt_swiglu_kernel<<<dim3(BSZ * NHEAD), blk, 0, stream>>>(qkv, w1, w2, xcat);

  // 3) 3x3 depthwise TTT branch -> xcat[:, 768:832]
  ttt_dwconv_kernel<<<dim3(BSZ * DDIM), blk, 0, stream>>>(qkv, w3, xcat);

  // 4) out = xcat @ proj_w^T + proj_b
  gemm_bias_kernel<<<dim3(CDIM / 64, M / 128), blk, 0, stream>>>(
      xcat, proj_w, proj_b, out, M, CDIM, XCATW);
}
